// TreeEncoder_77154792506027
// MI455X (gfx1250) — matrix operand order repaired and checked by reading, not compiled, not run on hardware
//
#include <hip/hip_runtime.h>
#include <math.h>

typedef _Float16 h16;
typedef h16   v16h __attribute__((ext_vector_type(16)));
typedef float v8f  __attribute__((ext_vector_type(8)));

// ---------------------------------------------------------------------------
// WMMA fragment layouts (CDNA5 ISA 7.12.2, wave32):
// A 16x32 f16: lanes 0-15 hold K {0..7,16..23}, lanes 16-31 hold K {8..15,24..31}
//              -> row-major A gives two contiguous 16B runs per lane (b128 x2).
// B 32x16 f16: lane lh*16+lm holds K = lh*8+e (e<8), 16+lh*8+e-8 (e>=8) for column lm (same as A).
//              -> we PRE-PACK B into [kt][nt][lane][e] so each lane reads 32
//                 contiguous bytes (b128 x2), fully coalesced across the wave.
// C/D 16x16 f32: lane&15 = N col; VGPR r = row r + 8*(lane>>4).
// ---------------------------------------------------------------------------
template <bool A_IS_INT>
__device__ inline v16h load_a_frag(const void* Av, int mrow, int k0, int lh, size_t lda) {
  v16h a;
  if constexpr (A_IS_INT) {
    const int* Ar = (const int*)Av + (size_t)mrow * lda + k0 + lh * 8;
#pragma unroll
    for (int v = 0; v < 8; ++v) a[v] = (h16)(float)Ar[v];
#pragma unroll
    for (int v = 0; v < 8; ++v) a[8 + v] = (h16)(float)Ar[16 + v];
  } else {
    const h16* Ar = (const h16*)Av + (size_t)mrow * lda + k0 + lh * 8;
#pragma unroll
    for (int v = 0; v < 8; ++v) a[v] = Ar[v];
#pragma unroll
    for (int v = 0; v < 8; ++v) a[8 + v] = Ar[16 + v];
  }
  return a;
}

// ---------------------------------------------------------------------------
// Generic GEMM: C[M,N](f32) = A[M,K] x Bpacked[K,N] (+C if accumulate)
// A: f16 row-major (A_IS_INT=false) or int32 0/1 mask (A_IS_INT=true).
// B: f16 pre-packed into WMMA fragment tiles (see pack_b_kernel).
// Workgroup: 256 threads = 8 waves arranged 2(rows) x 4(cols).
// Each wave: 32x64 C tile = 8 accumulators; per 32-K step: 2 A frags + 4 B
// frags (12 x b128 loads) feed 8 v_wmma_f32_16x16x32_f16.
// Requires M%64==0, N%256==0, K%32==0 (true for all calls below).
// ---------------------------------------------------------------------------
template <bool A_IS_INT>
__global__ __launch_bounds__(256)
void gemm_wmma_kernel(const void* __restrict__ Av, const h16* __restrict__ Bp,
                      float* __restrict__ C, int M, int N, int K,
                      int lda, int ldb, int ldc, int accumulate) {
  const int lane = threadIdx.x & 31;
  const int wave = threadIdx.x >> 5;
  const int wr = wave >> 2;        // 0..1
  const int wc = wave & 3;         // 0..3
  const int m0 = blockIdx.y * 64 + wr * 32;
  const int n0 = blockIdx.x * 256 + wc * 64;
  const int lm = lane & 15;
  const int lh = lane >> 4;
  const int nt0 = n0 >> 4;         // first of this wave's 4 column tiles
  const int ntiles = ldb >> 4;     // B column tiles per K-tile row

  v8f acc[2][4] = {};
  if (accumulate) {
#pragma unroll
    for (int i = 0; i < 2; ++i)
#pragma unroll
      for (int j = 0; j < 4; ++j) {
        size_t rb = (size_t)(m0 + 16 * i + 8 * lh) * (size_t)ldc + n0 + 16 * j + lm;
#pragma unroll
        for (int r = 0; r < 8; ++r) acc[i][j][r] = C[rb + (size_t)r * ldc];
      }
  }

  for (int k0 = 0; k0 < K; k0 += 32) {
    const int kt = k0 >> 5;
    v16h a0 = load_a_frag<A_IS_INT>(Av, m0 + lm,      k0, lh, (size_t)lda);
    v16h a1 = load_a_frag<A_IS_INT>(Av, m0 + 16 + lm, k0, lh, (size_t)lda);
    v16h b[4];
#pragma unroll
    for (int j = 0; j < 4; ++j) {
      const h16* p = Bp + (((size_t)kt * ntiles + nt0 + j) * 32 + lane) * 16;
#pragma unroll
      for (int e = 0; e < 16; ++e) b[j][e] = p[e];   // 32 contiguous bytes
    }
#pragma unroll
    for (int j = 0; j < 4; ++j) {
      acc[0][j] = __builtin_amdgcn_wmma_f32_16x16x32_f16(false, a0, false, b[j], (short)0, acc[0][j], false, false);
      acc[1][j] = __builtin_amdgcn_wmma_f32_16x16x32_f16(false, a1, false, b[j], (short)0, acc[1][j], false, false);
    }
  }

#pragma unroll
  for (int i = 0; i < 2; ++i)
#pragma unroll
    for (int j = 0; j < 4; ++j) {
      size_t rb = (size_t)(m0 + 16 * i + 8 * lh) * (size_t)ldc + n0 + 16 * j + lm;
#pragma unroll
      for (int r = 0; r < 8; ++r) C[rb + (size_t)r * ldc] = acc[i][j][r];
    }
}

// ---------------------------------------------------------------------------
// Pack f32 B[K,N] row-major -> f16 WMMA fragment tiles:
// dst[((kt*N/16 + nt)*32 + lh*16 + lm)*16 + e] = B[kt*32 + (e<8 ? lh*8 + e : 16 + lh*8 + e-8)][nt*16 + lm]
// ---------------------------------------------------------------------------
__global__ void pack_b_kernel(const float* __restrict__ src, h16* __restrict__ dst,
                              int K, int N) {
  int i = blockIdx.x * blockDim.x + threadIdx.x;
  if (i >= K * N) return;
  int k = i / N, n = i - k * N;
  int kt = k >> 5, kr = k & 31;
  int lh = (kr >> 3) & 1, e = (kr & 7) + ((kr >> 4) << 3);
  int nt = n >> 4, lm = n & 15;
  size_t di = (((size_t)kt * (N >> 4) + nt) * 32 + lh * 16 + lm) * 16 + e;
  dst[di] = (h16)src[i];
}

// ---------------------------------------------------------------------------
// Elementwise / reduction kernels
// ---------------------------------------------------------------------------
__global__ void fill_f32_kernel(float* __restrict__ p, float v, int n) {
  int i = blockIdx.x * blockDim.x + threadIdx.x;
  if (i < n) p[i] = v;
}

// Gather rel_E rows for timestep t into f16 [N,H]
__global__ void gather_rel_kernel(const float* __restrict__ relE, const int* __restrict__ paths,
                                  h16* __restrict__ Xt, int t, int Nrows, int S, int H) {
  int i = blockIdx.x * blockDim.x + threadIdx.x;
  if (i >= Nrows * H) return;
  int n = i / H, h = i - n * H;
  int idx = paths[n * S + t];
  Xt[i] = (h16)relE[(size_t)idx * H + h];
}

// LSTM cell: G = x@Wih + h@Whh (f32, [N,4H], gate order i,f,g,o); c in/out f32; h out f16
__global__ void lstm_cell_kernel(const float* __restrict__ G, const float* __restrict__ bih,
                                 const float* __restrict__ bhh, float* __restrict__ c,
                                 h16* __restrict__ hout, int Nrows, int H) {
  int i = blockIdx.x * blockDim.x + threadIdx.x;
  if (i >= Nrows * H) return;
  int n = i / H, h = i - n * H;
  const float* g = G + (size_t)n * 4 * H;
  float gi = g[h]         + bih[h]         + bhh[h];
  float gf = g[H + h]     + bih[H + h]     + bhh[H + h];
  float gg = g[2 * H + h] + bih[2 * H + h] + bhh[2 * H + h];
  float go = g[3 * H + h] + bih[3 * H + h] + bhh[3 * H + h];
  float si = 1.f / (1.f + expf(-gi));
  float sf = 1.f / (1.f + expf(-gf));
  float so = 1.f / (1.f + expf(-go));
  float cc = sf * c[i] + si * tanhf(gg);
  c[i] = cc;
  hout[i] = (h16)(so * tanhf(cc));
}

// y = act(in + bias) * scale; mode 0=none,1=relu,2=leaky(0.01); dual f32/f16 out
__global__ void bias_act_kernel(const float* __restrict__ in, const float* __restrict__ bias,
                                float* __restrict__ o32, h16* __restrict__ o16,
                                int total, int H, int mode, float scale) {
  int i = blockIdx.x * blockDim.x + threadIdx.x;
  if (i >= total) return;
  float v = in[i] + bias[i & (H - 1)];
  if (mode == 1)      v = v > 0.f ? v : 0.f;
  else if (mode == 2) v = v > 0.f ? v : 0.01f * v;
  v *= scale;
  o32[i] = v;
  o16[i] = (h16)v;
}

// mean over P inner rows: out[b,h] = mean_p in[(b*P+p), h]
__global__ void tree_mean_kernel(const float* __restrict__ in, float* __restrict__ out,
                                 int P, int H, int total) {
  int i = blockIdx.x * blockDim.x + threadIdx.x;
  if (i >= total) return;
  int b = i / H, h = i - b * H;
  float s = 0.f;
  for (int p = 0; p < P; ++p) s += in[(size_t)(b * P + p) * H + h];
  out[i] = s / (float)P;
}

// out[b,h] = mean_p ( pe[(b*P+p),h] / max(cnt[b*P+p],1) )
__global__ void entity_mean_kernel(const float* __restrict__ pe, const float* __restrict__ cnt,
                                   float* __restrict__ out, int P, int H, int total) {
  int i = blockIdx.x * blockDim.x + threadIdx.x;
  if (i >= total) return;
  int b = i / H, h = i - b * H;
  float s = 0.f;
  for (int p = 0; p < P; ++p) {
    int r = b * P + p;
    s += pe[(size_t)r * H + h] / fmaxf(cnt[r], 1.f);
  }
  out[i] = s / (float)P;
}

// per-row popcount of int 0/1 mask
__global__ __launch_bounds__(256)
void row_count_kernel(const int* __restrict__ m, float* __restrict__ cnt, int cols) {
  __shared__ float red[256];
  int row = blockIdx.x, tid = threadIdx.x;
  float s = 0.f;
  for (int c = tid; c < cols; c += 256) s += (float)m[(size_t)row * cols + c];
  red[tid] = s; __syncthreads();
  for (int off = 128; off; off >>= 1) { if (tid < off) red[tid] += red[tid + off]; __syncthreads(); }
  if (tid == 0) cnt[row] = red[0];
}

// LayerNorm over rows of length H (H==1024): y = (x[+addend]-m)/sqrt(v+eps)*g+b
__global__ __launch_bounds__(256)
void ln_rows_kernel(const float* __restrict__ x, const float* __restrict__ addend,
                    const float* __restrict__ g, const float* __restrict__ b,
                    float* __restrict__ y32, h16* __restrict__ y16, int H) {
  __shared__ float red[256];
  const int row = blockIdx.x, tid = threadIdx.x;
  const int per = H / 256;  // 4
  float xv[4];
  float s = 0.f;
  for (int j = 0; j < per; ++j) {
    int h = tid + j * 256;
    float v = x[(size_t)row * H + h];
    if (addend) v += addend[(size_t)row * H + h];
    xv[j] = v; s += v;
  }
  red[tid] = s; __syncthreads();
  for (int off = 128; off; off >>= 1) { if (tid < off) red[tid] += red[tid + off]; __syncthreads(); }
  float mean = red[0] / (float)H; __syncthreads();
  float q = 0.f;
  for (int j = 0; j < per; ++j) { float d = xv[j] - mean; q += d * d; }
  red[tid] = q; __syncthreads();
  for (int off = 128; off; off >>= 1) { if (tid < off) red[tid] += red[tid + off]; __syncthreads(); }
  float inv = rsqrtf(red[0] / (float)H + 1e-5f);
  for (int j = 0; j < per; ++j) {
    int h = tid + j * 256;
    float v = (xv[j] - mean) * inv * g[h] + b[h];
    y32[(size_t)row * H + h] = v;
    y16[(size_t)row * H + h] = (h16)v;
  }
}

// row L2-normalize, writing two copies (ws + final output)
__global__ __launch_bounds__(256)
void normalize_rows_kernel(const float* __restrict__ p, float* __restrict__ o1,
                           float* __restrict__ o2, int H) {
  __shared__ float red[256];
  int row = blockIdx.x, tid = threadIdx.x;
  float s = 0.f;
  for (int h = tid; h < H; h += 256) { float v = p[(size_t)row * H + h]; s += v * v; }
  red[tid] = s; __syncthreads();
  for (int off = 128; off; off >>= 1) { if (tid < off) red[tid] += red[tid + off]; __syncthreads(); }
  float inv = 1.f / fmaxf(sqrtf(red[0]), 1e-12f);
  for (int h = tid; h < H; h += 256) {
    float v = p[(size_t)row * H + h] * inv;
    o1[(size_t)row * H + h] = v;
    o2[(size_t)row * H + h] = v;
  }
}

// sim[i,j] = (p_i . p_j - (i==j)) / tao   (one block per row i, 128 threads)
__global__ __launch_bounds__(128)
void sim_kernel(const float* __restrict__ p, float* __restrict__ sim, int Bn, int H, float invtao) {
  __shared__ float pi[1024];
  int i = blockIdx.x;
  for (int h = threadIdx.x; h < H; h += 128) pi[h] = p[(size_t)i * H + h];
  __syncthreads();
  int j = threadIdx.x;
  float d = 0.f;
  for (int h = 0; h < H; ++h) d += pi[h] * p[(size_t)j * H + h];
  if (i == j) d -= 1.f;
  sim[i * Bn + j] = d * invtao;
}

// loss = mean_i -log_softmax(sim[i])[y_i], y = i+1-(i%2)*2
__global__ __launch_bounds__(128)
void loss_kernel(const float* __restrict__ sim, float* __restrict__ out, int Bn) {
  __shared__ float red[128];
  int i = threadIdx.x;
  const float* row = sim + (size_t)i * Bn;
  float mx = -3.4e38f;
  for (int j = 0; j < Bn; ++j) mx = fmaxf(mx, row[j]);
  float se = 0.f;
  for (int j = 0; j < Bn; ++j) se += expf(row[j] - mx);
  int y = i + 1 - (i & 1) * 2;
  red[i] = -(row[y] - mx - logf(se));
  __syncthreads();
  for (int off = 64; off; off >>= 1) { if (i < off) red[i] += red[i + off]; __syncthreads(); }
  if (i == 0) out[0] = red[0] / (float)Bn;
}

// ---------------------------------------------------------------------------
// Host orchestration
// ---------------------------------------------------------------------------
extern "C" void kernel_launch(void* const* d_in, const int* in_sizes, int n_in,
                              void* d_out, int out_size, void* d_ws, size_t ws_size,
                              hipStream_t stream) {
  (void)in_sizes; (void)n_in; (void)out_size; (void)ws_size;
  const int H = 1024, Bn = 128, Pn = 16, Sn = 12, Nn = Bn * Pn, H4 = 4 * H, ENT = 20000;
  const float BN_SCALE = 1.0f / sqrtf(1.0f + 1e-5f);

  const float* rel_E = (const float*)d_in[0];
  const float* Wih = (const float*)d_in[1];
  const float* Whh = (const float*)d_in[2];
  const float* bih = (const float*)d_in[3];
  const float* bhh = (const float*)d_in[4];
  const float* Wm1 = (const float*)d_in[5];
  const float* bm1 = (const float*)d_in[6];
  const float* Wm2 = (const float*)d_in[7];
  const float* bm2 = (const float*)d_in[8];
  const float* ent_E = (const float*)d_in[9];
  const float* Wf1 = (const float*)d_in[10];
  const float* bf1 = (const float*)d_in[11];
  const float* Wf2 = (const float*)d_in[12];
  const float* bf2 = (const float*)d_in[13];
  const float* g1e = (const float*)d_in[14];
  const float* b1e = (const float*)d_in[15];
  const float* g2e = (const float*)d_in[16];
  const float* b2e = (const float*)d_in[17];
  // d_in[18..21] = Wq,bq,Wk,bk (unused: singleton-key attention collapses)
  const float* Wv = (const float*)d_in[22];
  const float* bv = (const float*)d_in[23];
  const float* Wo = (const float*)d_in[24];
  const float* bo = (const float*)d_in[25];
  const float* g1t = (const float*)d_in[26];
  const float* b1t = (const float*)d_in[27];
  const float* Wp1 = (const float*)d_in[28];
  const float* bp1 = (const float*)d_in[29];
  const float* Wp2 = (const float*)d_in[30];
  const float* bp2 = (const float*)d_in[31];
  const int* paths = (const int*)d_in[32];
  const int* entities = (const int*)d_in[33];
  float* out = (float*)d_out;

  // ---- workspace carve-out ----
  char* base = (char*)d_ws;
  size_t off = 0;
  auto alloc = [&](size_t bytes) -> void* {
    void* p = base + off;
    off += (bytes + 255) & ~(size_t)255;
    return p;
  };
  h16* wihp  = (h16*)alloc((size_t)H * H4 * 2);     // packed B tiles
  h16* whhp  = (h16*)alloc((size_t)H * H4 * 2);
  h16* wm1p  = (h16*)alloc((size_t)2 * H * H * 2);
  h16* wm2p  = (h16*)alloc((size_t)H * H * 2);
  h16* entEp = (h16*)alloc((size_t)ENT * H * 2);
  h16* wf1p  = (h16*)alloc((size_t)H * H * 2);
  h16* wf2p  = (h16*)alloc((size_t)H * H * 2);
  h16* wvp   = (h16*)alloc((size_t)H * H * 2);
  h16* wop   = (h16*)alloc((size_t)H * H * 2);
  h16* wp1p  = (h16*)alloc((size_t)H * H * 2);
  h16* wp2p  = (h16*)alloc((size_t)H * H * 2);
  h16* xt16  = (h16*)alloc((size_t)Nn * H * 2);
  h16* hs16  = (h16*)alloc((size_t)Sn * Nn * H * 2);
  float* G    = (float*)alloc((size_t)Nn * H4 * 4);  // LSTM gates; chunks reused in rec phase
  float* G2   = G + (size_t)Nn * H;
  float* cbuf = (float*)alloc((size_t)Nn * H * 4);
  float* a32  = (float*)alloc((size_t)Nn * H * 4);
  h16*   a16  = (h16*)alloc((size_t)Nn * H * 2);
  float* e32  = (float*)alloc((size_t)Nn * H * 4);
  h16*   e16  = (h16*)alloc((size_t)Nn * H * 2);
  float* pe32 = (float*)alloc((size_t)Nn * H * 4);
  float* cnt  = (float*)alloc((size_t)Nn * 4);
  float* eent = (float*)alloc((size_t)Bn * H * 4);
  float* treeb = (float*)alloc((size_t)Bn * H * 4);
  float* sbuf  = (float*)alloc((size_t)Bn * H * 4);
  float* pA32 = (float*)alloc((size_t)Bn * H * 4);  h16* pA16 = (h16*)alloc((size_t)Bn * H * 2);
  float* pB32 = (float*)alloc((size_t)Bn * H * 4);  h16* pB16 = (h16*)alloc((size_t)Bn * H * 2);
  float* pC32 = (float*)alloc((size_t)Bn * H * 4);  h16* pC16 = (h16*)alloc((size_t)Bn * H * 2);
  float* pD32 = (float*)alloc((size_t)Bn * H * 4);  h16* pD16 = (h16*)alloc((size_t)Bn * H * 2);
  float* pn32 = (float*)alloc((size_t)Bn * H * 4);
  float* simb = (float*)alloc((size_t)Bn * Bn * 4);

  auto packB = [&](const float* s, h16* d, int K, int N) {
    size_t n = (size_t)K * N;
    pack_b_kernel<<<(unsigned)((n + 255) / 256), 256, 0, stream>>>(s, d, K, N);
  };
  auto gemmF = [&](const h16* A, const h16* Bm, float* C, int M, int Nc, int K,
                   int lda, int ldb, int ldc, int acc) {
    dim3 g(Nc / 256, M / 64);
    gemm_wmma_kernel<false><<<g, 256, 0, stream>>>((const void*)A, Bm, C, M, Nc, K, lda, ldb, ldc, acc);
  };
  auto gemmI = [&](const int* A, const h16* Bm, float* C, int M, int Nc, int K,
                   int lda, int ldb, int ldc, int acc) {
    dim3 g(Nc / 256, M / 64);
    gemm_wmma_kernel<true><<<g, 256, 0, stream>>>((const void*)A, Bm, C, M, Nc, K, lda, ldb, ldc, acc);
  };
  const int NE = Nn * H;             // 2M elements
  const int BH = Bn * H;             // 128K elements
  const unsigned gNE = (NE + 255) / 256, gBH = (BH + 255) / 256;

  // ---- weight conversion f32 -> packed f16 B tiles (L2-resident across reuse) ----
  packB(Wih, wihp, H, H4);
  packB(Whh, whhp, H, H4);
  packB(Wm1, wm1p, 2 * H, H);    // kt-major: bottom half at offset H*H
  packB(Wm2, wm2p, H, H);
  packB(ent_E, entEp, ENT, H);
  packB(Wf1, wf1p, H, H);
  packB(Wf2, wf2p, H, H);
  packB(Wv, wvp, H, H);
  packB(Wo, wop, H, H);
  packB(Wp1, wp1p, H, H);
  packB(Wp2, wp2p, H, H);

  // ---- PathEncoder: LSTM over S=12 steps ----
  fill_f32_kernel<<<gNE, 256, 0, stream>>>(cbuf, 0.f, NE);
  for (int t = 0; t < Sn; ++t) {
    gather_rel_kernel<<<gNE, 256, 0, stream>>>(rel_E, paths, xt16, t, Nn, Sn, H);
    gemmF(xt16, wihp, G, Nn, H4, H, H, H4, H4, 0);
    if (t > 0)
      gemmF(hs16 + (size_t)(t - 1) * Nn * H, whhp, G, Nn, H4, H, H, H4, H4, 1);
    lstm_cell_kernel<<<gNE, 256, 0, stream>>>(G, bih, bhh, cbuf, hs16 + (size_t)t * Nn * H, Nn, H);
  }

  // ---- recursive MLP over hs[1..11]; concat handled as split-K accumulation ----
  const h16* ecur = hs16;  // hs[0]
  for (int t = 1; t < Sn; ++t) {
    gemmF(ecur, wm1p, G, Nn, H, H, H, H, H, 0);                                   // e part
    gemmF(hs16 + (size_t)t * Nn * H, wm1p + (size_t)H * H, G, Nn, H, H, H, H, H, 1);  // ht part
    bias_act_kernel<<<gNE, 256, 0, stream>>>(G, bm1, a32, a16, NE, H, 2, 1.f);
    gemmF(a16, wm2p, G2, Nn, H, H, H, H, H, 0);
    bias_act_kernel<<<gNE, 256, 0, stream>>>(G2, bm2, e32, e16, NE, H, 2, 1.f);
    ecur = e16;
  }
  tree_mean_kernel<<<gBH, 256, 0, stream>>>(e32, treeb, Pn, H, BH);

  // ---- EntityEncoder: mask GEMM (int A on the fly) + pooled MLP ----
  row_count_kernel<<<Nn, 256, 0, stream>>>(entities, cnt, ENT);
  gemmI(entities, entEp, pe32, Nn, H, ENT, ENT, H, H, 0);
  entity_mean_kernel<<<gBH, 256, 0, stream>>>(pe32, cnt, eent, Pn, H, BH);
  ln_rows_kernel<<<Bn, 256, 0, stream>>>(eent, nullptr, g1e, b1e, pA32, pA16, H);
  gemmF(pA16, wf1p, sbuf, Bn, H, H, H, H, H, 0);
  bias_act_kernel<<<gBH, 256, 0, stream>>>(sbuf, bf1, pB32, pB16, BH, H, 1, BN_SCALE);
  ln_rows_kernel<<<Bn, 256, 0, stream>>>(pB32, nullptr, g2e, b2e, pC32, pC16, H);
  gemmF(pC16, wf2p, sbuf, Bn, H, H, H, H, H, 0);
  bias_act_kernel<<<gBH, 256, 0, stream>>>(sbuf, bf2, pD32, pD16, BH, H, 1, BN_SCALE);

  // ---- collapsed cross-attention: attn = (e@Wv+bv)@Wo+bo ----
  gemmF(pD16, wvp, sbuf, Bn, H, H, H, H, H, 0);
  bias_act_kernel<<<gBH, 256, 0, stream>>>(sbuf, bv, pA32, pA16, BH, H, 0, 1.f);
  gemmF(pA16, wop, sbuf, Bn, H, H, H, H, H, 0);
  bias_act_kernel<<<gBH, 256, 0, stream>>>(sbuf, bo, pB32, pB16, BH, H, 0, 1.f);
  ln_rows_kernel<<<Bn, 256, 0, stream>>>(treeb, pB32, g1t, b1t, pC32, pC16, H);

  // ---- projection head + normalize ----
  gemmF(pC16, wp1p, sbuf, Bn, H, H, H, H, H, 0);
  bias_act_kernel<<<gBH, 256, 0, stream>>>(sbuf, bp1, pA32, pA16, BH, H, 1, 1.f);
  gemmF(pA16, wp2p, sbuf, Bn, H, H, H, H, H, 0);
  bias_act_kernel<<<gBH, 256, 0, stream>>>(sbuf, bp2, pB32, pB16, BH, H, 0, 1.f);
  normalize_rows_kernel<<<Bn, 256, 0, stream>>>(pB32, pn32, out + 1, H);

  // ---- contrastive loss ----
  sim_kernel<<<Bn, 128, 0, stream>>>(pn32, simb, Bn, H, 1.f / 0.5f);
  loss_kernel<<<1, 128, 0, stream>>>(simb, out, Bn);
}
